// YakAttention_50079318672052
// MI455X (gfx1250) — matrix operand order repaired and checked by reading, not compiled, not run on hardware
//
#include <hip/hip_runtime.h>
#include <hip/hip_bf16.h>

// ---------------------------------------------------------------------------
// MI455X (gfx1250) attention block: QKV GEMM -> RoPE -> causal GQA flash
// attention -> O-proj.  All matmuls via v_wmma_f32_16x16x32_bf16 (wave32).
// Weights are converted+transposed to bf16 [N][K] once so GEMM LDS staging is
// pure b128 traffic; V is stored d-major so attention needs no LDS transpose.
// ---------------------------------------------------------------------------

typedef __bf16 bf16;
typedef __attribute__((ext_vector_type(16))) __bf16 v16bf;
typedef __attribute__((ext_vector_type(8)))  __bf16 bf16x8;
typedef __attribute__((ext_vector_type(8)))  float  v8f;

union Frag16 { v16bf v; bf16x8 h[2]; };

#define B_    2
#define S_    2048
#define H_    4096
#define NH_   32
#define NKV_  8
#define HD_   128
#define QKV_N 6144
#define MTOT  4096   // B_*S_

__device__ __forceinline__ v8f wmma_bf(v16bf a, v16bf b, v8f c) {
  return __builtin_amdgcn_wmma_f32_16x16x32_bf16(false, a, false, b, (short)0, c,
                                                 false, false);
}

// ---------------------------------------------------------------------------
// 1a) f32 -> bf16 conversion (activations)
// ---------------------------------------------------------------------------
__global__ __launch_bounds__(256) void cvt_f32_bf16(const float* __restrict__ in,
                                                    bf16* __restrict__ out, int n) {
  int i = blockIdx.x * blockDim.x + threadIdx.x;
  if (i < n) out[i] = (bf16)in[i];
}

// ---------------------------------------------------------------------------
// 1b) f32 [R][C] -> bf16 transposed [C][R]  (weights), 32x32 LDS tiles
// ---------------------------------------------------------------------------
__global__ __launch_bounds__(256) void cvt_transpose_f32_bf16(
    const float* __restrict__ in, bf16* __restrict__ out, int R, int C) {
  __shared__ float tile[32][33];
  const int c0 = blockIdx.x * 32, r0 = blockIdx.y * 32;
  const int tx = threadIdx.x & 31, ty = threadIdx.x >> 5;  // 32 x 8
#pragma unroll
  for (int i = 0; i < 32; i += 8)
    tile[ty + i][tx] = in[(size_t)(r0 + ty + i) * C + c0 + tx];
  __syncthreads();
#pragma unroll
  for (int i = 0; i < 32; i += 8)
    out[(size_t)(c0 + ty + i) * R + r0 + tx] = (bf16)tile[tx][ty + i];
}

// ---------------------------------------------------------------------------
// 2) QKV GEMM:  C[4096 x 6144] = hs[4096 x 4096] @ w_qkv^T stored [6144][4096]
//    Block tile 128(M) x 64(N), K-step 32.  8 waves, each 2x2 of 16x16.
//    Epilogue scatters bf16 into Q [B,NH,S,HD], K [B,NKV,S,HD],
//    V transposed [B,NKV,HD,S].
// ---------------------------------------------------------------------------
__global__ __launch_bounds__(256) void qkv_gemm_kernel(const bf16* __restrict__ A,
                                                       const bf16* __restrict__ WT,
                                                       bf16* __restrict__ Q,
                                                       bf16* __restrict__ K,
                                                       bf16* __restrict__ Vt) {
  __shared__ __align__(16) bf16 As[128][32];  // [m][k]
  __shared__ __align__(16) bf16 Bs[64][32];   // [n][k]

  const int tid  = threadIdx.x;
  const int lane = tid & 31;
  const int wave = tid >> 5;
  const int wm   = wave >> 1;        // 0..3
  const int wn   = wave & 1;         // 0..1
  const int mBase = blockIdx.y * 128;
  const int nBase = blockIdx.x * 64;
  const int lrow = lane & 15;
  const int half = lane >> 4;

  const bf16* aSrc = A  + (size_t)(mBase + (tid >> 1)) * H_ + (tid & 1) * 16;
  const bf16* bSrc = WT + (size_t)(nBase + (tid >> 2)) * H_ + (tid & 3) * 8;

  v8f acc[2][2] = {};

  for (int k0 = 0; k0 < H_; k0 += 32) {
    { // A tile: 128x32, 2 threads per row, 16 bf16 each
      int r = tid >> 1, ks = (tid & 1) * 16;
      *(bf16x8*)&As[r][ks]     = *(const bf16x8*)(aSrc + k0);
      *(bf16x8*)&As[r][ks + 8] = *(const bf16x8*)(aSrc + k0 + 8);
    }
    { // B tile: 64(n) x 32(k) from transposed weights, 4 threads per row
      int n = tid >> 2, ks = (tid & 3) * 8;
      *(bf16x8*)&Bs[n][ks] = *(const bf16x8*)(bSrc + k0);
    }
    if (k0 + 32 < H_) {  // prefetch next K-step (global_prefetch_b8)
      __builtin_prefetch(aSrc + k0 + 32, 0, 1);
      __builtin_prefetch(bSrc + k0 + 32, 0, 1);
    }
    __syncthreads();

    Frag16 af[2], bf[2];
#pragma unroll
    for (int sm = 0; sm < 2; ++sm) {
      const bf16* p = &As[wm * 32 + sm * 16 + lrow][half * 8];
      af[sm].h[0] = *(const bf16x8*)p;
      af[sm].h[1] = *(const bf16x8*)(p + 16);
    }
#pragma unroll
    for (int sn = 0; sn < 2; ++sn) {
      const bf16* p = &Bs[wn * 32 + sn * 16 + lrow][half * 8];
      bf[sn].h[0] = *(const bf16x8*)p;
      bf[sn].h[1] = *(const bf16x8*)(p + 16);
    }
#pragma unroll
    for (int sm = 0; sm < 2; ++sm)
#pragma unroll
      for (int sn = 0; sn < 2; ++sn)
        acc[sm][sn] = wmma_bf(af[sm].v, bf[sn].v, acc[sm][sn]);
    __syncthreads();
  }

  // Scatter epilogue: C layout row = r + 8*half, col = lane%16
#pragma unroll
  for (int sm = 0; sm < 2; ++sm)
#pragma unroll
    for (int sn = 0; sn < 2; ++sn) {
      int col = nBase + wn * 32 + sn * 16 + lrow;
#pragma unroll
      for (int r = 0; r < 8; ++r) {
        int row = mBase + wm * 32 + sm * 16 + half * 8 + r;
        int b = row >> 11, s = row & (S_ - 1);
        bf16 val = (bf16)acc[sm][sn][r];
        if (col < NH_ * HD_) {
          int hh = col >> 7, d = col & 127;
          Q[(((size_t)b * NH_ + hh) * S_ + s) * HD_ + d] = val;
        } else if (col < NH_ * HD_ + NKV_ * HD_) {
          int c = col - NH_ * HD_;
          int hh = c >> 7, d = c & 127;
          K[(((size_t)b * NKV_ + hh) * S_ + s) * HD_ + d] = val;
        } else {
          int c = col - (NH_ * HD_ + NKV_ * HD_);
          int hh = c >> 7, d = c & 127;
          Vt[(((size_t)b * NKV_ + hh) * HD_ + d) * S_ + s] = val;  // d-major
        }
      }
    }
}

// ---------------------------------------------------------------------------
// 3) NeoX RoPE in place on Q and K (thread owns pair (d, d+64))
// ---------------------------------------------------------------------------
__global__ __launch_bounds__(256) void rope_kernel(const int* __restrict__ positions,
                                                   bf16* __restrict__ Q,
                                                   bf16* __restrict__ K) {
  const size_t nq = (size_t)B_ * NH_ * S_ * 64;
  const size_t nk = (size_t)B_ * NKV_ * S_ * 64;
  size_t i = (size_t)blockIdx.x * blockDim.x + threadIdx.x;
  bf16* ptr; int nh; size_t idx;
  if (i < nq)            { ptr = Q; nh = NH_;  idx = i; }
  else if (i < nq + nk)  { ptr = K; nh = NKV_; idx = i - nq; }
  else return;

  int d = (int)(idx & 63); idx >>= 6;
  int s = (int)(idx % S_); idx /= S_;
  int h = (int)(idx % nh);
  int b = (int)(idx / nh);
  int pos = positions[(size_t)b * S_ + s];

  float inv = __powf(10000.0f, -(float)d * (1.0f / 64.0f));
  float ang = (float)pos * inv;
  float c = __cosf(ang), sn = __sinf(ang);

  size_t base = (((size_t)b * nh + h) * S_ + s) * HD_ + d;
  float x1 = (float)ptr[base];
  float x2 = (float)ptr[base + 64];
  ptr[base]      = (bf16)(x1 * c - x2 * sn);
  ptr[base + 64] = (bf16)(x2 * c + x1 * sn);
}

// ---------------------------------------------------------------------------
// 4) Causal GQA flash attention.
//    grid = (S/64, NH, B), block = 128 (4 waves x 16 query rows).
//    K in [B,NKV,S,HD]; V pre-transposed in [B,NKV,HD,S].
// ---------------------------------------------------------------------------
__global__ __launch_bounds__(128) void attn_kernel(const bf16* __restrict__ Q,
                                                   const bf16* __restrict__ K,
                                                   const bf16* __restrict__ Vg,
                                                   bf16* __restrict__ Out) {
  __shared__ __align__(16) bf16 Ks[64][128];   // [key][d]
  __shared__ __align__(16) bf16 Vs[128][64];   // [d][key]
  __shared__ __align__(16) bf16 Ps[4][16][64]; // per-wave P staging

  const int qt = blockIdx.x, h = blockIdx.y, b = blockIdx.z;
  const int hkv = h >> 2;                      // NH/NKV = 4
  const int tid = threadIdx.x;
  const int lane = tid & 31;
  const int wave = tid >> 5;
  const int lrow = lane & 15;
  const int half = lane >> 4;
  const int qBase = qt * 64;
  const int qRow0 = qBase + wave * 16;

  const bf16* Qb  = Q  + ((size_t)b * NH_ + h) * S_ * HD_;
  const bf16* Kb  = K  + ((size_t)b * NKV_ + hkv) * S_ * HD_;
  const bf16* Vtb = Vg + ((size_t)b * NKV_ + hkv) * HD_ * S_;  // [d][S]

  // Preload Q fragments (A layout), 4 d-chunks of 32
  Frag16 qf[4];
  {
    const bf16* qrow = Qb + (size_t)(qRow0 + lrow) * HD_ + half * 8;
#pragma unroll
    for (int dc = 0; dc < 4; ++dc) {
      qf[dc].h[0] = *(const bf16x8*)(qrow + dc * 32);
      qf[dc].h[1] = *(const bf16x8*)(qrow + dc * 32 + 16);
    }
  }

  v8f o[8] = {};
  float m_run[8], l_run[8];
#pragma unroll
  for (int r = 0; r < 8; ++r) { m_run[r] = -3.0e38f; l_run[r] = 0.0f; }

  const float scale = 0.08838834764831845f; // 1/sqrt(128)

  for (int j0 = 0; j0 <= qBase; j0 += 64) {
    { // K tile: [64][128], 2 threads per key row
      int key = tid >> 1, ds0 = (tid & 1) * 64;
      const bf16* src = Kb + (size_t)(j0 + key) * HD_ + ds0;
#pragma unroll
      for (int j = 0; j < 64; j += 8)
        *(bf16x8*)&Ks[key][ds0 + j] = *(const bf16x8*)(src + j);
    }
    { // V tile: [128][64] straight from d-major global, 1 thread per d row
      const bf16* src = Vtb + (size_t)tid * S_ + j0;
#pragma unroll
      for (int j = 0; j < 64; j += 8)
        *(bf16x8*)&Vs[tid][j] = *(const bf16x8*)(src + j);
    }
    __syncthreads();

    // S = Q K^T : 4 subtiles of 16 keys, K-dim = 128 in chunks of 32
    v8f st[4];
#pragma unroll
    for (int t = 0; t < 4; ++t) {
      v8f acc = {};
#pragma unroll
      for (int dc = 0; dc < 4; ++dc) {
        const bf16* p = &Ks[t * 16 + lrow][dc * 32 + half * 8];
        Frag16 kf;
        kf.h[0] = *(const bf16x8*)p;
        kf.h[1] = *(const bf16x8*)(p + 16);
        acc = wmma_bf(qf[dc].v, kf.v, acc);
      }
      st[t] = acc;
    }

    // scale + causal mask + new row max
    float mnew[8];
#pragma unroll
    for (int r = 0; r < 8; ++r) mnew[r] = m_run[r];
#pragma unroll
    for (int t = 0; t < 4; ++t) {
      int key = j0 + t * 16 + lrow;
#pragma unroll
      for (int r = 0; r < 8; ++r) {
        int q = qRow0 + half * 8 + r;
        float vv = st[t][r] * scale;
        vv = (key <= q) ? vv : -3.0e38f;
        st[t][r] = vv;
        mnew[r] = fmaxf(mnew[r], vv);
      }
    }
#pragma unroll
    for (int off = 1; off < 16; off <<= 1)
#pragma unroll
      for (int r = 0; r < 8; ++r)
        mnew[r] = fmaxf(mnew[r], __shfl_xor(mnew[r], off, 32));

    // rescale running state, exponentiate, row sums
    float rsum[8];
#pragma unroll
    for (int r = 0; r < 8; ++r) {
      float alpha = __expf(m_run[r] - mnew[r]);
      l_run[r] *= alpha;
#pragma unroll
      for (int dn = 0; dn < 8; ++dn) o[dn][r] *= alpha;
      m_run[r] = mnew[r];
      rsum[r] = 0.0f;
    }
#pragma unroll
    for (int t = 0; t < 4; ++t)
#pragma unroll
      for (int r = 0; r < 8; ++r) {
        float p = __expf(st[t][r] - mnew[r]);
        st[t][r] = p;
        rsum[r] += p;
      }
#pragma unroll
    for (int off = 1; off < 16; off <<= 1)
#pragma unroll
      for (int r = 0; r < 8; ++r) rsum[r] += __shfl_xor(rsum[r], off, 32);
#pragma unroll
    for (int r = 0; r < 8; ++r) l_run[r] += rsum[r];

    // stage P through per-wave LDS to get A-fragment layout
#pragma unroll
    for (int t = 0; t < 4; ++t)
#pragma unroll
      for (int r = 0; r < 8; ++r)
        Ps[wave][half * 8 + r][t * 16 + lrow] = (bf16)st[t][r];

    // O += P @ V  (key chunks of 32, 8 output d-subtiles)
#pragma unroll
    for (int kc = 0; kc < 2; ++kc) {
      const bf16* prow = &Ps[wave][lrow][kc * 32 + half * 8];
      Frag16 pf;
      pf.h[0] = *(const bf16x8*)prow;
      pf.h[1] = *(const bf16x8*)(prow + 16);
#pragma unroll
      for (int dn = 0; dn < 8; ++dn) {
        const bf16* vrow = &Vs[dn * 16 + lrow][kc * 32 + half * 8];
        Frag16 vf;
        vf.h[0] = *(const bf16x8*)vrow;
        vf.h[1] = *(const bf16x8*)(vrow + 16);
        o[dn] = wmma_bf(pf.v, vf.v, o[dn]);
      }
    }
    __syncthreads();
  }

  // normalize + store bf16 into [B,S,NH*HD]
#pragma unroll
  for (int dn = 0; dn < 8; ++dn)
#pragma unroll
    for (int r = 0; r < 8; ++r) {
      int s = qRow0 + half * 8 + r;
      int d = dn * 16 + lrow;
      float vv = o[dn][r] / l_run[r];
      Out[((size_t)b * S_ + s) * H_ + h * HD_ + d] = (bf16)vv;
    }
}

// ---------------------------------------------------------------------------
// 5) O-projection:  out[4096 x 4096] = attn[4096 x 4096] @ w_o^T stored
//    [4096(N)][4096(K)].  Same tiling as QKV GEMM, f32 output to d_out.
// ---------------------------------------------------------------------------
__global__ __launch_bounds__(256) void oproj_gemm_kernel(const bf16* __restrict__ A,
                                                         const bf16* __restrict__ WT,
                                                         float* __restrict__ C) {
  __shared__ __align__(16) bf16 As[128][32];
  __shared__ __align__(16) bf16 Bs[64][32];

  const int tid  = threadIdx.x;
  const int lane = tid & 31;
  const int wave = tid >> 5;
  const int wm   = wave >> 1;
  const int wn   = wave & 1;
  const int mBase = blockIdx.y * 128;
  const int nBase = blockIdx.x * 64;
  const int lrow = lane & 15;
  const int half = lane >> 4;

  const bf16* aSrc = A  + (size_t)(mBase + (tid >> 1)) * H_ + (tid & 1) * 16;
  const bf16* bSrc = WT + (size_t)(nBase + (tid >> 2)) * H_ + (tid & 3) * 8;

  v8f acc[2][2] = {};

  for (int k0 = 0; k0 < H_; k0 += 32) {
    {
      int r = tid >> 1, ks = (tid & 1) * 16;
      *(bf16x8*)&As[r][ks]     = *(const bf16x8*)(aSrc + k0);
      *(bf16x8*)&As[r][ks + 8] = *(const bf16x8*)(aSrc + k0 + 8);
    }
    {
      int n = tid >> 2, ks = (tid & 3) * 8;
      *(bf16x8*)&Bs[n][ks] = *(const bf16x8*)(bSrc + k0);
    }
    if (k0 + 32 < H_) {
      __builtin_prefetch(aSrc + k0 + 32, 0, 1);
      __builtin_prefetch(bSrc + k0 + 32, 0, 1);
    }
    __syncthreads();

    Frag16 af[2], bf[2];
#pragma unroll
    for (int sm = 0; sm < 2; ++sm) {
      const bf16* p = &As[wm * 32 + sm * 16 + lrow][half * 8];
      af[sm].h[0] = *(const bf16x8*)p;
      af[sm].h[1] = *(const bf16x8*)(p + 16);
    }
#pragma unroll
    for (int sn = 0; sn < 2; ++sn) {
      const bf16* p = &Bs[wn * 32 + sn * 16 + lrow][half * 8];
      bf[sn].h[0] = *(const bf16x8*)p;
      bf[sn].h[1] = *(const bf16x8*)(p + 16);
    }
#pragma unroll
    for (int sm = 0; sm < 2; ++sm)
#pragma unroll
      for (int sn = 0; sn < 2; ++sn)
        acc[sm][sn] = wmma_bf(af[sm].v, bf[sn].v, acc[sm][sn]);
    __syncthreads();
  }

#pragma unroll
  for (int sm = 0; sm < 2; ++sm)
#pragma unroll
    for (int sn = 0; sn < 2; ++sn) {
      int col = nBase + wn * 32 + sn * 16 + lrow;
#pragma unroll
      for (int r = 0; r < 8; ++r) {
        int row = mBase + wm * 32 + sm * 16 + half * 8 + r;
        C[(size_t)row * H_ + col] = acc[sm][sn][r];
      }
    }
}

// ---------------------------------------------------------------------------
// Host launcher
// ---------------------------------------------------------------------------
extern "C" void kernel_launch(void* const* d_in, const int* in_sizes, int n_in,
                              void* d_out, int out_size, void* d_ws, size_t ws_size,
                              hipStream_t stream) {
  const int*   positions = (const int*)d_in[0];
  const float* hidden    = (const float*)d_in[1];
  const float* w_qkv     = (const float*)d_in[2];
  const float* w_o       = (const float*)d_in[3];
  float* out = (float*)d_out;

  char* ws = (char*)d_ws;
  size_t off = 0;
  auto carve = [&](size_t bytes) -> void* {
    void* p = ws + off;
    off += (bytes + 255) & ~(size_t)255;
    return p;
  };
  bf16* hsB    = (bf16*)carve((size_t)MTOT * H_ * 2);            // 32 MB
  bf16* wqkvT  = (bf16*)carve((size_t)QKV_N * H_ * 2);           // 48 MB [N][K]
  bf16* woT    = (bf16*)carve((size_t)H_ * H_ * 2);              // 32 MB [N][K]
  bf16* Qb     = (bf16*)carve((size_t)B_ * NH_  * S_ * HD_ * 2); // 32 MB
  bf16* Kb     = (bf16*)carve((size_t)B_ * NKV_ * S_ * HD_ * 2); //  8 MB
  bf16* Vtb    = (bf16*)carve((size_t)B_ * NKV_ * HD_ * S_ * 2); //  8 MB [d][S]
  bf16* attnB  = (bf16*)carve((size_t)MTOT * H_ * 2);            // 32 MB

  // 1) convert activations; convert + transpose weights to [N][K]
  cvt_f32_bf16<<<(MTOT * H_) / 256, 256, 0, stream>>>(hidden, hsB, MTOT * H_);
  cvt_transpose_f32_bf16<<<dim3(QKV_N / 32, H_ / 32), 256, 0, stream>>>(
      w_qkv, wqkvT, H_, QKV_N);
  cvt_transpose_f32_bf16<<<dim3(H_ / 32, H_ / 32), 256, 0, stream>>>(
      w_o, woT, H_, H_);

  // 2) QKV projection (WMMA); V scattered d-major
  qkv_gemm_kernel<<<dim3(QKV_N / 64, MTOT / 128), 256, 0, stream>>>(
      hsB, wqkvT, Qb, Kb, Vtb);

  // 3) RoPE in place on Q and K
  {
    size_t total = (size_t)B_ * NH_ * S_ * 64 + (size_t)B_ * NKV_ * S_ * 64;
    rope_kernel<<<(unsigned)((total + 255) / 256), 256, 0, stream>>>(positions, Qb, Kb);
  }

  // 4) causal GQA flash attention (WMMA)
  attn_kernel<<<dim3(S_ / 64, NH_, B_), 128, 0, stream>>>(Qb, Kb, Vtb, attnB);

  // 5) output projection (WMMA), f32 straight to d_out
  oproj_gemm_kernel<<<dim3(H_ / 64, MTOT / 128), 256, 0, stream>>>(attnB, woT, out);
}
